// SlotAttention_44702019617367
// MI455X (gfx1250) — compile-verified, matrix operand order checked by reading, not run on hardware
//
#include <hip/hip_runtime.h>
#include <hip/hip_bf16.h>

typedef __attribute__((ext_vector_type(16))) _Float16 v16h;
typedef __attribute__((ext_vector_type(8)))  _Float16 v8h;
typedef __attribute__((ext_vector_type(8)))  float    v8f;

#define B_    32
#define N_    4096
#define D_    256
#define S_    8
#define ITERS 3

// ---------------------------------------------------------------------------
// WMMA helpers (CDNA5 16x16x32 f16 -> f32)
// ---------------------------------------------------------------------------
static __device__ __forceinline__ v8f wmma_f16(v16h a, v16h b, v8f c) {
  // 8 args: (neg_a, A, neg_b, B, c_mod, C, reuse_a, reuse_b)
  return __builtin_amdgcn_wmma_f32_16x16x32_f16(false, a, false, b, (short)0, c,
                                                false, false);
}

// Load one 16x32 f16 fragment (A-style, row-major source).
// Layout per ISA 7.12.2: lane L (0-15) row M=L, halves 0..7 = K kb..kb+7,
// halves 8..15 = K kb+16..kb+23 where kb = (L>>4)*8.
// Works identically for the B matrix with "row" = output column n
// (source stored [n][k] row-major), assuming the mirrored striping.
static __device__ __forceinline__ v16h load_frag(const _Float16* __restrict__ base,
                                                 int ld, int r0, int k0) {
  int lane = threadIdx.x & 31;
  int r  = r0 + (lane & 15);
  int kb = k0 + ((lane >> 4) << 3);
  const _Float16* p = base + (size_t)r * ld + kb;
  v8h lo = *(const v8h*)(p);        // 16B contiguous
  v8h hi = *(const v8h*)(p + 16);   // 16B contiguous
  return __builtin_shufflevector(lo, hi, 0, 1, 2, 3, 4, 5, 6, 7,
                                 8, 9, 10, 11, 12, 13, 14, 15);
}

// ---------------------------------------------------------------------------
// Stage 0: f32 -> f16 weight conversion
// ---------------------------------------------------------------------------
__global__ void f32_to_f16_kernel(const float* __restrict__ s,
                                  _Float16* __restrict__ d, int n) {
  int i = blockIdx.x * blockDim.x + threadIdx.x;
  if (i < n) d[i] = (_Float16)s[i];
}

// ---------------------------------------------------------------------------
// Stage 1: LayerNorm(inputs) -> x_h (f16).  One wave per row of 256.
// ---------------------------------------------------------------------------
__global__ void ln_x_kernel(const float* __restrict__ x,
                            const float* __restrict__ w,
                            const float* __restrict__ bb,
                            _Float16* __restrict__ out) {
  int wave = (int)((blockIdx.x * blockDim.x + threadIdx.x) >> 5);
  int lane = threadIdx.x & 31;
  const float* row = x + (size_t)wave * D_;
  float v[8];
  float s = 0.f, sq = 0.f;
#pragma unroll
  for (int i = 0; i < 8; ++i) {
    v[i] = row[lane * 8 + i];
    s += v[i];
    sq += v[i] * v[i];
  }
  for (int off = 1; off < 32; off <<= 1) {
    s  += __shfl_xor(s, off, 32);
    sq += __shfl_xor(sq, off, 32);
  }
  float mean = s * (1.0f / D_);
  float var  = sq * (1.0f / D_) - mean * mean;
  float inv  = rsqrtf(var + 1e-5f);
  v8h o;
#pragma unroll
  for (int i = 0; i < 8; ++i) {
    int d = lane * 8 + i;
    o[i] = (_Float16)((v[i] - mean) * inv * w[d] + bb[d]);
  }
  *(v8h*)(out + (size_t)wave * D_ + lane * 8) = o;
}

// ---------------------------------------------------------------------------
// Stage 2: fused k/v projection.  Wave computes one 16x16 tile of BOTH
// k = x@k_w.T + k_b and v = x@v_w.T + v_b (A fragment shared).
// Writes k as [b][token][d] (f16) and v transposed as [b][d][token] (f16).
// ---------------------------------------------------------------------------
__global__ void kv_gemm_kernel(const _Float16* __restrict__ x,
                               const _Float16* __restrict__ kw,
                               const _Float16* __restrict__ vw,
                               const float* __restrict__ kb,
                               const float* __restrict__ vb,
                               _Float16* __restrict__ kh,
                               _Float16* __restrict__ vT) {
  int wave = (int)((blockIdx.x * blockDim.x + threadIdx.x) >> 5);
  int lane = threadIdx.x & 31;
  int mt = wave >> 4;   // 0..8191 token tiles
  int nt = wave & 15;   // 0..15 feature tiles
  v8f ck = {0, 0, 0, 0, 0, 0, 0, 0};
  v8f cv = {0, 0, 0, 0, 0, 0, 0, 0};
#pragma unroll
  for (int ks = 0; ks < 8; ++ks) {
    v16h a   = load_frag(x,  D_, mt * 16, ks * 32);
    v16h bkf = load_frag(kw, D_, nt * 16, ks * 32);
    v16h bvf = load_frag(vw, D_, nt * 16, ks * 32);
    ck = wmma_f16(a, bkf, ck);
    cv = wmma_f16(a, bvf, cv);
  }
  int n     = nt * 16 + (lane & 15);
  int mbase = mt * 16 + ((lane >> 4) << 3);
  float kbias = kb[n], vbias = vb[n];
#pragma unroll
  for (int r = 0; r < 8; ++r) {
    int m = mbase + r;
    kh[(size_t)m * D_ + n] = (_Float16)(ck[r] + kbias);
    int b = m >> 12;            // 4096 tokens per batch
    int j = m & (N_ - 1);
    vT[((size_t)b * D_ + n) * N_ + j] = (_Float16)(cv[r] + vbias);
  }
}

// ---------------------------------------------------------------------------
// Stage 3: slots init (broadcast slots_mu over batch)
// ---------------------------------------------------------------------------
__global__ void init_slots_kernel(const float* __restrict__ mu,
                                  float* __restrict__ slots) {
  int i = blockIdx.x * blockDim.x + threadIdx.x;  // 32*8*256
  slots[i] = mu[i & (S_ * D_ - 1)];
}

__global__ void zero_kernel(float* __restrict__ p, int n) {
  int i = blockIdx.x * blockDim.x + threadIdx.x;
  if (i < n) p[i] = 0.f;
}

// ---------------------------------------------------------------------------
// Stage 4: q = LN(slots)@q_w.T + q_b -> q_h (f16, padded to 16 slot rows).
// One 256-thread block per (b, padded-slot) row.
// ---------------------------------------------------------------------------
__global__ void q_proj_kernel(const float* __restrict__ slots,
                              const float* __restrict__ lnw,
                              const float* __restrict__ lnb,
                              const float* __restrict__ qw,
                              const float* __restrict__ qb,
                              _Float16* __restrict__ qh) {
  __shared__ float sx[D_];
  __shared__ float red[16];
  __shared__ float stat[2];
  int rb = blockIdx.x;  // b*16 + s_pad
  int sp = rb & 15;
  int b  = rb >> 4;
  int t  = threadIdx.x;
  if (sp >= S_) {  // zero pad rows (block-uniform branch)
    qh[(size_t)rb * D_ + t] = (_Float16)0.0f;
    return;
  }
  const float* row = slots + ((size_t)b * S_ + sp) * D_;
  float x = row[t];
  float ss = x, sq = x * x;
  for (int off = 16; off; off >>= 1) {
    ss += __shfl_xor(ss, off, 32);
    sq += __shfl_xor(sq, off, 32);
  }
  int wid = t >> 5;
  if ((t & 31) == 0) { red[wid] = ss; red[8 + wid] = sq; }
  __syncthreads();
  if (t == 0) {
    float a = 0.f, b2 = 0.f;
    for (int i = 0; i < 8; ++i) { a += red[i]; b2 += red[8 + i]; }
    float mean = a / D_;
    float var  = b2 / D_ - mean * mean;
    stat[0] = mean;
    stat[1] = rsqrtf(var + 1e-5f);
  }
  __syncthreads();
  sx[t] = (x - stat[0]) * stat[1] * lnw[t] + lnb[t];
  __syncthreads();
  float acc = qb[t];
  const float* wr = qw + (size_t)t * D_;
  for (int d = 0; d < D_; ++d) acc += sx[d] * wr[d];
  qh[(size_t)rb * D_ + t] = (_Float16)acc;
}

// ---------------------------------------------------------------------------
// Stage 5: dots = q@kT * scale, fused softmax over the SLOT axis.
// Per C/D layout, lane L (0-15) holds all 8 slot values for its token column
// in c[0..7] -> slot softmax is a per-lane 8-element reduction.
// Writes attn_slot (+EPS) to output, accumulates per-(b,slot) row sums.
// ---------------------------------------------------------------------------
__global__ void dots_softmax_kernel(const _Float16* __restrict__ qh,
                                    const _Float16* __restrict__ kh,
                                    float* __restrict__ attn_slot_out,
                                    float* __restrict__ rowsum) {
  int wave = (int)((blockIdx.x * blockDim.x + threadIdx.x) >> 5);
  int lane = threadIdx.x & 31;
  int b  = wave >> 8;   // 256 token tiles per batch
  int nt = wave & 255;
  const _Float16* qb_ = qh + (size_t)b * 16 * D_;
  const _Float16* kb_ = kh + (size_t)b * N_ * D_;
  v8f c = {0, 0, 0, 0, 0, 0, 0, 0};
#pragma unroll
  for (int ks = 0; ks < 8; ++ks) {
    v16h a  = load_frag(qb_, D_, 0, ks * 32);
    v16h bf = load_frag(kb_, D_, nt * 16, ks * 32);
    c = wmma_f16(a, bf, c);
  }
  const float scale = 0.0625f;  // 256^-0.5
  float av[8];
  float mx = -1e30f;
#pragma unroll
  for (int i = 0; i < 8; ++i) { av[i] = c[i] * scale; mx = fmaxf(mx, av[i]); }
  float se = 0.f;
#pragma unroll
  for (int i = 0; i < 8; ++i) { av[i] = __expf(av[i] - mx); se += av[i]; }
  float inv = 1.0f / se;
#pragma unroll
  for (int i = 0; i < 8; ++i) av[i] = av[i] * inv + 1e-8f;
  int j = nt * 16 + lane;
  if (lane < 16) {
#pragma unroll
    for (int i = 0; i < 8; ++i)
      attn_slot_out[((size_t)b * S_ + i) * N_ + j] = av[i];
  }
#pragma unroll
  for (int i = 0; i < 8; ++i) {
    float v = (lane < 16) ? av[i] : 0.f;
    for (int off = 1; off < 16; off <<= 1) v += __shfl_xor(v, off, 32);
    if (lane == 0) atomicAdd(&rowsum[b * S_ + i], v);
  }
}

// ---------------------------------------------------------------------------
// Stage 6: attn = attn_slot / rowsum; write f32 output + f16 (padded) copy.
// ---------------------------------------------------------------------------
__global__ void attn_norm_kernel(const float* __restrict__ attn_slot,
                                 const float* __restrict__ rowsum,
                                 float* __restrict__ attn_out,
                                 _Float16* __restrict__ attn_h) {
  int tid = blockIdx.x * blockDim.x + threadIdx.x;  // B*16*N
  int j = tid & (N_ - 1);
  int i = (tid >> 12) & 15;
  int b = tid >> 16;
  if (i < S_) {
    size_t src = ((size_t)b * S_ + i) * N_ + j;
    float a = attn_slot[src] / rowsum[b * S_ + i];
    attn_out[src] = a;
    attn_h[tid] = (_Float16)a;
  } else {
    attn_h[tid] = (_Float16)0.0f;  // pad slot rows for WMMA
  }
}

// ---------------------------------------------------------------------------
// Stage 7: updates = attn @ v  via WMMA (A = attn_h [16 x 4096],
// B column n = feature d, contiguous K from vT [d][token]).
// ---------------------------------------------------------------------------
__global__ void updates_gemm_kernel(const _Float16* __restrict__ attn_h,
                                    const _Float16* __restrict__ vT,
                                    float* __restrict__ updates) {
  int wave = (int)((blockIdx.x * blockDim.x + threadIdx.x) >> 5);
  int lane = threadIdx.x & 31;
  int b  = wave >> 4;
  int dt = wave & 15;
  const _Float16* ab  = attn_h + (size_t)b * 16 * N_;
  const _Float16* vb_ = vT + (size_t)b * D_ * N_;
  v8f c = {0, 0, 0, 0, 0, 0, 0, 0};
  for (int ks = 0; ks < N_ / 32; ++ks) {
    v16h a  = load_frag(ab,  N_, 0, ks * 32);
    v16h bf = load_frag(vb_, N_, dt * 16, ks * 32);
    c = wmma_f16(a, bf, c);
  }
  if (lane < 16) {
    int d = dt * 16 + lane;
#pragma unroll
    for (int i = 0; i < 8; ++i)
      updates[((size_t)b * S_ + i) * D_ + d] = c[i];
  }
}

// ---------------------------------------------------------------------------
// Stage 8: GRUCell + LN + FF residual.  One 256-thread block per (b,slot).
// Small (256 rows), weights are L2-resident -> plain f32 VALU for accuracy.
// ---------------------------------------------------------------------------
__global__ void gru_ff_kernel(const float* __restrict__ updates,
                              float* __restrict__ slots,
                              const float* __restrict__ wih,
                              const float* __restrict__ whh,
                              const float* __restrict__ bih,
                              const float* __restrict__ bhh,
                              const float* __restrict__ lnw,
                              const float* __restrict__ lnb,
                              const float* __restrict__ fc1w,
                              const float* __restrict__ fc1b,
                              const float* __restrict__ fc2w,
                              const float* __restrict__ fc2b,
                              float* __restrict__ out_slots, int write_out) {
  __shared__ float s_u[D_], s_h[D_], s_ff[D_], s_hid[D_];
  __shared__ float red[16], stat[2];
  int rb = blockIdx.x;
  int t  = threadIdx.x;
  size_t base = (size_t)rb * D_;
  s_u[t] = updates[base + t];
  s_h[t] = slots[base + t];
  __syncthreads();
  float gi[3], gh[3];
#pragma unroll
  for (int g = 0; g < 3; ++g) {
    int row = g * D_ + t;
    const float* wi = wih + (size_t)row * D_;
    const float* wh = whh + (size_t)row * D_;
    float a = bih[row], c2 = bhh[row];
    for (int d = 0; d < D_; ++d) { a += s_u[d] * wi[d]; c2 += s_h[d] * wh[d]; }
    gi[g] = a;
    gh[g] = c2;
  }
  float r  = 1.f / (1.f + __expf(-(gi[0] + gh[0])));
  float z  = 1.f / (1.f + __expf(-(gi[1] + gh[1])));
  float ng = tanhf(gi[2] + r * gh[2]);
  float ns = (1.f - z) * ng + z * s_h[t];
  // LayerNorm(ns) across the block
  float ss = ns, sq = ns * ns;
  for (int off = 16; off; off >>= 1) {
    ss += __shfl_xor(ss, off, 32);
    sq += __shfl_xor(sq, off, 32);
  }
  if ((t & 31) == 0) { red[t >> 5] = ss; red[8 + (t >> 5)] = sq; }
  __syncthreads();
  if (t == 0) {
    float a = 0.f, b2 = 0.f;
    for (int i = 0; i < 8; ++i) { a += red[i]; b2 += red[8 + i]; }
    float mean = a / D_;
    float var  = b2 / D_ - mean * mean;
    stat[0] = mean;
    stat[1] = rsqrtf(var + 1e-5f);
  }
  __syncthreads();
  s_ff[t] = (ns - stat[0]) * stat[1] * lnw[t] + lnb[t];
  __syncthreads();
  float h1 = fc1b[t];
  const float* w1 = fc1w + (size_t)t * D_;
  for (int d = 0; d < D_; ++d) h1 += s_ff[d] * w1[d];
  s_hid[t] = fmaxf(h1, 0.f);
  __syncthreads();
  float o = ns + fc2b[t];
  const float* w2 = fc2w + (size_t)t * D_;  // HID == D_
  for (int d = 0; d < D_; ++d) o += s_hid[d] * w2[d];
  slots[base + t] = o;
  if (write_out) out_slots[base + t] = o;
}

// ---------------------------------------------------------------------------
extern "C" void kernel_launch(void* const* d_in, const int* in_sizes, int n_in,
                              void* d_out, int out_size, void* d_ws, size_t ws_size,
                              hipStream_t stream) {
  (void)in_sizes; (void)n_in; (void)out_size; (void)ws_size;
  const float* inputs   = (const float*)d_in[0];
  const float* slots_mu = (const float*)d_in[1];
  const float* q_w = (const float*)d_in[2];
  const float* q_b = (const float*)d_in[3];
  const float* k_w = (const float*)d_in[4];
  const float* k_b = (const float*)d_in[5];
  const float* v_w = (const float*)d_in[6];
  const float* v_b = (const float*)d_in[7];
  const float* gru_w_ih = (const float*)d_in[8];
  const float* gru_w_hh = (const float*)d_in[9];
  const float* gru_b_ih = (const float*)d_in[10];
  const float* gru_b_hh = (const float*)d_in[11];
  const float* fc1_w = (const float*)d_in[12];
  const float* fc1_b = (const float*)d_in[13];
  const float* fc2_w = (const float*)d_in[14];
  const float* fc2_b = (const float*)d_in[15];
  const float* ln_in_w = (const float*)d_in[16];
  const float* ln_in_b = (const float*)d_in[17];
  const float* ln_sl_w = (const float*)d_in[18];
  const float* ln_sl_b = (const float*)d_in[19];
  const float* ln_ff_w = (const float*)d_in[20];
  const float* ln_ff_b = (const float*)d_in[21];

  char* ws = (char*)d_ws;
  _Float16* x_h    = (_Float16*)(ws + 0);          // 64 MB   [B*N][D]
  _Float16* k_h    = (_Float16*)(ws + 67108864);   // 64 MB   [B][N][D]
  _Float16* vT_h   = (_Float16*)(ws + 134217728);  // 64 MB   [B][D][N]
  _Float16* kw_h   = (_Float16*)(ws + 201326592);  // 128 KB
  _Float16* vw_h   = (_Float16*)(ws + 201457664);  // 128 KB
  _Float16* q_h    = (_Float16*)(ws + 201588736);  // 256 KB  [B][16][D]
  _Float16* attn_h = (_Float16*)(ws + 201850880);  // 4 MB    [B][16][N]
  float* rowsum    = (float*)(ws + 206045184);     // 1 KB    [B][S]
  float* slots     = (float*)(ws + 206046208);     // 256 KB  [B][S][D]
  float* updates   = (float*)(ws + 206308352);     // 256 KB  [B][S][D]

  float* out       = (float*)d_out;
  float* out_slots = out;                    // [B][S][D]      65536
  float* out_attn  = out + 65536;            // [3][B][S][N]   3145728
  float* out_aslot = out + 65536 + 3145728;  // [3][B][S][N]   3145728

  f32_to_f16_kernel<<<256, 256, 0, stream>>>(k_w, kw_h, D_ * D_);
  f32_to_f16_kernel<<<256, 256, 0, stream>>>(v_w, vw_h, D_ * D_);
  ln_x_kernel<<<16384, 256, 0, stream>>>(inputs, ln_in_w, ln_in_b, x_h);
  kv_gemm_kernel<<<16384, 256, 0, stream>>>(x_h, kw_h, vw_h, k_b, v_b, k_h, vT_h);
  init_slots_kernel<<<256, 256, 0, stream>>>(slots_mu, slots);

  for (int iter = 0; iter < ITERS; ++iter) {
    float* as_iter = out_aslot + (size_t)iter * 1048576;
    float* at_iter = out_attn + (size_t)iter * 1048576;
    zero_kernel<<<1, 256, 0, stream>>>(rowsum, B_ * S_);
    q_proj_kernel<<<512, 256, 0, stream>>>(slots, ln_sl_w, ln_sl_b, q_w, q_b, q_h);
    dots_softmax_kernel<<<1024, 256, 0, stream>>>(q_h, k_h, as_iter, rowsum);
    attn_norm_kernel<<<8192, 256, 0, stream>>>(as_iter, rowsum, at_iter, attn_h);
    updates_gemm_kernel<<<64, 256, 0, stream>>>(attn_h, vT_h, updates);
    gru_ff_kernel<<<256, 256, 0, stream>>>(updates, slots, gru_w_ih, gru_w_hh,
                                           gru_b_ih, gru_b_hh, ln_ff_w, ln_ff_b,
                                           fc1_w, fc1_b, fc2_w, fc2_b,
                                           out_slots, iter == 2 ? 1 : 0);
  }
}
